// BERTModel_38079180046919
// MI455X (gfx1250) — matrix operand order repaired and checked by reading, not compiled, not run on hardware
//
#include <hip/hip_runtime.h>

// ---------------------------------------------------------------------------
// BERT forward on gfx1250 (MI455X). All GEMMs via v_wmma_f32_16x16x32_bf16.
// B=8 T=512 C=768 H=12 D=64 L=12 V=32000. All GEMM dims are multiples of the
// 128x64 tile / 32 K-chunk -> no edge handling.
//
// v3: double-buffered LDS, one-barrier software pipeline (prefetch chunk i+1
// while 8 WMMAs run on chunk i), per-thread staging pointers bumped by a
// constant stride (no per-iteration 64-bit muls). Weights/activations are
// pre-converted to bf16 once per launch. Workspace use ~452 MB.
// ---------------------------------------------------------------------------

#define Bc 8
#define Tc 512
#define Cc 768
#define Hc 12
#define Lc 12
#define Dc 64
#define Vc 32000
#define BT (Bc * Tc)      // 4096
#define FC (4 * Cc)       // 3072
#define LDP 40            // padded LDS row (bf16): 80B = 5*16B -> aligned, conflict-free

typedef __attribute__((ext_vector_type(16))) __bf16 v16bf;
typedef __attribute__((ext_vector_type(8)))  __bf16 v8bf;
typedef __attribute__((ext_vector_type(8)))  float  v8f;

__device__ __forceinline__ __bf16 f2bf(float f) {
    union { float f; unsigned u; } in; in.f = f;
    unsigned r = in.u + 0x7FFFu + ((in.u >> 16) & 1u);   // round-to-nearest-even
    unsigned short h = (unsigned short)(r >> 16);
    union { unsigned short u; __bf16 b; } out; out.u = h;
    return out.b;
}

// ---------------------------------------------------------------------------
// Batched bf16 GEMM: C = scale*(A @ B) + bias, optional ReLU.
// A: bf16 [M,K] row-major (lda). B: bf16, bmode:
//   0 = row-major [K,N] (ldb)                      (weights, smV)
//   1 = transposed: elem(k,n) = B[n*ldb + k]       (Q @ K^T)
//   2 = QKV [H,C,D]: N-tile (64) pins head h=n0/64 -> row-major with ldb=64
// Outputs: Cf (f32) and/or Cb (bf16), same layout (ldc), either may be null.
// Batch z -> (zb,zh) = (z/nH, z%nH) with per-operand strides.
// Block 128 = 4 waves; tile 128x64; wave = 32x64 strip; 8 WMMA / K-chunk.
// Pipeline: store regs->lds[cur]; barrier; prefetch next; compute lds[cur].
// ---------------------------------------------------------------------------
__global__ __launch_bounds__(128)
void gemm_wmma_bf16(const __bf16* __restrict__ A, long lda,
                    const __bf16* __restrict__ B, long ldb, int bmode,
                    const float* __restrict__ bias, float scale, int relu,
                    float* __restrict__ Cf, __bf16* __restrict__ Cb, long ldc,
                    int K, int nH,
                    long sAb, long sAh, long sBb, long sBh,
                    long sCb, long sCh)
{
    __shared__ __bf16 lA[2][128][LDP];   // A tiles, row-major [M][K]
    __shared__ __bf16 lBt[2][64][LDP];   // B tiles, TRANSPOSED [N][K]

    const int tid = threadIdx.x;
    const int zb = blockIdx.z / nH, zh = blockIdx.z % nH;
    A += zb * sAb + zh * sAh;
    B += zb * sBb + zh * sBh;
    const long coff = zb * sCb + zh * sCh;

    const int m0 = blockIdx.y * 128;
    const int n0 = blockIdx.x * 64;
    if (bmode == 2) B += (long)(n0 >> 6) * ((long)K << 6);  // head base, ldb==64
    const int nB = (bmode == 2) ? 0 : n0;

    const int wave  = tid >> 5;
    const int lane  = tid & 31;
    const int l16   = lane & 15;
    const int lhalf = lane >> 4;

    // --- per-thread staging coordinates + pointers (constant strides) ------
    int rA[4], cA[4];
    const __bf16* aptr[4];
    #pragma unroll
    for (int j = 0; j < 4; ++j) {
        int c = tid + j * 128;
        rA[j] = c >> 2;
        cA[j] = (c & 3) << 3;
        aptr[j] = A + (long)(m0 + rA[j]) * lda + cA[j];
    }
    int bn[2], bk[2];
    const __bf16* bptr[2];
    long bstep;
    if (bmode == 1) {                // global already [N,K]: contiguous K
        #pragma unroll
        for (int j = 0; j < 2; ++j) {
            int c = tid + j * 128;
            bn[j] = c >> 2;          // nn
            bk[j] = (c & 3) << 3;    // k8
            bptr[j] = B + (long)(n0 + bn[j]) * ldb + bk[j];
        }
        bstep = 32;
    } else {                         // row-major [K,N]: read along n, scatter
        #pragma unroll
        for (int j = 0; j < 2; ++j) {
            int c = tid + j * 128;
            bk[j] = c >> 3;          // kk
            bn[j] = (c & 7) << 3;    // n8
            bptr[j] = B + (long)bk[j] * ldb + nB + bn[j];
        }
        bstep = 32 * ldb;
    }

    v8f acc[2][4] = {};
    v8bf ra[4], rb[2];

    // --- prefetch chunk 0 --------------------------------------------------
    #pragma unroll
    for (int j = 0; j < 4; ++j) { ra[j] = *(const v8bf*)aptr[j]; aptr[j] += 32; }
    #pragma unroll
    for (int j = 0; j < 2; ++j) { rb[j] = *(const v8bf*)bptr[j]; bptr[j] += bstep; }

    const int nch = K >> 5;
    for (int it = 0; it < nch; ++it) {
        const int cur = it & 1;
        // --- store staged regs into lds[cur] ------------------------------
        #pragma unroll
        for (int j = 0; j < 4; ++j) *(v8bf*)&lA[cur][rA[j]][cA[j]] = ra[j];
        if (bmode == 1) {
            #pragma unroll
            for (int j = 0; j < 2; ++j) *(v8bf*)&lBt[cur][bn[j]][bk[j]] = rb[j];
        } else {
            #pragma unroll
            for (int j = 0; j < 2; ++j)
                #pragma unroll
                for (int p = 0; p < 8; ++p) lBt[cur][bn[j] + p][bk[j]] = rb[j][p];
        }
        __syncthreads();
        // --- issue prefetch for chunk it+1 (retires during WMMAs) ---------
        if (it + 1 < nch) {
            #pragma unroll
            for (int j = 0; j < 4; ++j) { ra[j] = *(const v8bf*)aptr[j]; aptr[j] += 32; }
            #pragma unroll
            for (int j = 0; j < 2; ++j) { rb[j] = *(const v8bf*)bptr[j]; bptr[j] += bstep; }
        }
        // --- compute from lds[cur] ----------------------------------------
        v16bf af[2];
        #pragma unroll
        for (int r2 = 0; r2 < 2; ++r2) {
            const int arow = (wave << 5) + (r2 << 4) + l16;
            v8bf a0 = *(const v8bf*)&lA[cur][arow][lhalf << 3];
            v8bf a1 = *(const v8bf*)&lA[cur][arow][16 + (lhalf << 3)];
            af[r2] = __builtin_shufflevector(a0, a1,
                0, 1, 2, 3, 4, 5, 6, 7, 8, 9, 10, 11, 12, 13, 14, 15);
        }
        #pragma unroll
        for (int s = 0; s < 4; ++s) {
            const int col = (s << 4) + l16;
            const __bf16* bp = &lBt[cur][col][lhalf << 3];
            v8bf b0 = *(const v8bf*)bp;
            v8bf b1 = *(const v8bf*)(bp + 16);
            v16bf bfrag = __builtin_shufflevector(b0, b1,
                0, 1, 2, 3, 4, 5, 6, 7, 8, 9, 10, 11, 12, 13, 14, 15);
            acc[0][s] = __builtin_amdgcn_wmma_f32_16x16x32_bf16(
                false, af[0], false, bfrag, (short)0, acc[0][s], false, false);
            acc[1][s] = __builtin_amdgcn_wmma_f32_16x16x32_bf16(
                false, af[1], false, bfrag, (short)0, acc[1][s], false, false);
        }
    }

    // --- epilogue (ISA C/D layout: VGPR r -> M=r (+8 for lane>=16), N=lane%16)
    #pragma unroll
    for (int r2 = 0; r2 < 2; ++r2) {
        #pragma unroll
        for (int s = 0; s < 4; ++s) {
            const int col = n0 + (s << 4) + l16;
            const float bv = bias ? bias[col] : 0.0f;
            #pragma unroll
            for (int r = 0; r < 8; ++r) {
                const int row = m0 + (wave << 5) + (r2 << 4) + (lhalf << 3) + r;
                float v = acc[r2][s][r] * scale + bv;
                if (relu) v = fmaxf(v, 0.0f);
                const long o = coff + (long)row * ldc + col;
                if (Cf) Cf[o] = v;
                if (Cb) Cb[o] = f2bf(v);
            }
        }
    }
}

// ---------------------------------------------------------------------------
__global__ __launch_bounds__(256)
void f32_to_bf16_kernel(const float* __restrict__ in, __bf16* __restrict__ o,
                        long n)
{
    long i = (long)blockIdx.x * 256 + threadIdx.x;
    const long stride = (long)gridDim.x * 256;
    for (; i < n; i += stride) o[i] = f2bf(in[i]);
}

// x[bt,:] = emb[ids[bt],:] + pos[t,:]  (f32 + bf16 copies)
__global__ __launch_bounds__(256)
void embed_kernel(const int* __restrict__ ids, const float* __restrict__ emb,
                  const float* __restrict__ pos,
                  float* __restrict__ x, __bf16* __restrict__ xb)
{
    const long bt = blockIdx.x;
    const int  t  = (int)(bt % Tc);
    const long id = ids[bt];
    for (int c = threadIdx.x; c < Cc; c += 256) {
        float v = emb[id * Cc + c] + pos[(long)t * Cc + c];
        x[bt * Cc + c]  = v;
        xb[bt * Cc + c] = f2bf(v);
    }
}

// Softmax over the QUERY axis (reference quirk). att f32 in, bf16 out.
__global__ __launch_bounds__(256)
void softmax_q_axis(const float* __restrict__ att, __bf16* __restrict__ attb)
{
    const int  k    = blockIdx.x * 256 + threadIdx.x;
    const long base = (long)blockIdx.y * Tc * Tc + k;
    float m = -3.4e38f;
    for (int q = 0; q < Tc; ++q) m = fmaxf(m, att[base + (long)q * Tc]);
    float ssum = 0.0f;
    for (int q = 0; q < Tc; ++q) ssum += __expf(att[base + (long)q * Tc] - m);
    const float inv = 1.0f / ssum;
    for (int q = 0; q < Tc; ++q) {
        const long o = base + (long)q * Tc;
        attb[o] = f2bf(__expf(att[o] - m) * inv);
    }
}

// out = LN(x + dx)*g + b  -> f32 out (in-place safe) + bf16 copy
__global__ __launch_bounds__(256)
void add_layernorm(const float* __restrict__ x, const float* __restrict__ dx,
                   const float* __restrict__ g, const float* __restrict__ b,
                   float* __restrict__ out, __bf16* __restrict__ outb)
{
    __shared__ float red[256];
    const long row = blockIdx.x;
    const float* xr = x  + row * Cc;
    const float* dr = dx + row * Cc;

    float v[3];
    float s = 0.0f;
    #pragma unroll
    for (int i = 0; i < 3; ++i) {
        int c = threadIdx.x + i * 256;
        v[i] = xr[c] + dr[c];
        s += v[i];
    }
    red[threadIdx.x] = s; __syncthreads();
    for (int w = 128; w > 0; w >>= 1) {
        if (threadIdx.x < w) red[threadIdx.x] += red[threadIdx.x + w];
        __syncthreads();
    }
    const float mu = red[0] * (1.0f / Cc);
    __syncthreads();

    float s2 = 0.0f;
    #pragma unroll
    for (int i = 0; i < 3; ++i) { float d = v[i] - mu; s2 += d * d; }
    red[threadIdx.x] = s2; __syncthreads();
    for (int w = 128; w > 0; w >>= 1) {
        if (threadIdx.x < w) red[threadIdx.x] += red[threadIdx.x + w];
        __syncthreads();
    }
    const float inv = rsqrtf(red[0] * (1.0f / Cc) + 1e-5f);

    #pragma unroll
    for (int i = 0; i < 3; ++i) {
        int c = threadIdx.x + i * 256;
        float o = (v[i] - mu) * inv * g[c] + b[c];
        out[row * Cc + c]  = o;
        outb[row * Cc + c] = f2bf(o);
    }
}

// ---------------------------------------------------------------------------
extern "C" void kernel_launch(void* const* d_in, const int* in_sizes, int n_in,
                              void* d_out, int out_size, void* d_ws, size_t ws_size,
                              hipStream_t stream)
{
    const int*   ids   = (const int*)  d_in[0];
    const float* emb   = (const float*)d_in[1];
    const float* pos   = (const float*)d_in[2];
    const float* Wq    = (const float*)d_in[3];
    const float* Wk    = (const float*)d_in[4];
    const float* Wv    = (const float*)d_in[5];
    const float* Wo    = (const float*)d_in[6];
    const float* bo    = (const float*)d_in[7];
    const float* ln1g  = (const float*)d_in[8];
    const float* ln1b  = (const float*)d_in[9];
    const float* W1    = (const float*)d_in[10];
    const float* b1    = (const float*)d_in[11];
    const float* W2    = (const float*)d_in[12];
    const float* b2    = (const float*)d_in[13];
    const float* ln2g  = (const float*)d_in[14];
    const float* ln2b  = (const float*)d_in[15];
    const float* decW  = (const float*)d_in[16];
    const float* decb  = (const float*)d_in[17];
    float* out = (float*)d_out;

    const long NBTC = (long)BT * Cc;              // 3.15M
    const long NATT = (long)Bc * Hc * Tc * Tc;    // 25.2M
    const long NQKV = (long)Lc * Hc * Cc * Dc;    // 7.08M
    const long NWO  = (long)Lc * Cc * Cc;
    const long NW1  = (long)Lc * Cc * FC;
    const long NW2  = (long)Lc * FC * Cc;
    const long NDEC = (long)Cc * Vc;

    // ---- workspace carve: f32 region, then bf16 region (~452 MB total)
    float* x    = (float*)d_ws;
    float* t1   = x  + NBTC;
    float* attf = t1 + NBTC;
    __bf16* bp  = (__bf16*)(attf + NATT);
    __bf16* xb   = bp; bp += NBTC;
    __bf16* Qb   = bp; bp += NBTC;
    __bf16* Kb   = bp; bp += NBTC;
    __bf16* Vb   = bp; bp += NBTC;
    __bf16* Ob   = bp; bp += NBTC;
    __bf16* h1b  = bp; bp += (long)BT * FC;
    __bf16* attb = bp; bp += NATT;
    __bf16* wqb  = bp; bp += NQKV;
    __bf16* wkb  = bp; bp += NQKV;
    __bf16* wvb  = bp; bp += NQKV;
    __bf16* wob  = bp; bp += NWO;
    __bf16* w1b  = bp; bp += NW1;
    __bf16* w2b  = bp; bp += NW2;
    __bf16* wdb  = bp; bp += NDEC;

    // ---- pre-convert all weights to bf16 (once per launch)
    f32_to_bf16_kernel<<<2048, 256, 0, stream>>>(Wq,   wqb, NQKV);
    f32_to_bf16_kernel<<<2048, 256, 0, stream>>>(Wk,   wkb, NQKV);
    f32_to_bf16_kernel<<<2048, 256, 0, stream>>>(Wv,   wvb, NQKV);
    f32_to_bf16_kernel<<<2048, 256, 0, stream>>>(Wo,   wob, NWO);
    f32_to_bf16_kernel<<<2048, 256, 0, stream>>>(W1,   w1b, NW1);
    f32_to_bf16_kernel<<<2048, 256, 0, stream>>>(W2,   w2b, NW2);
    f32_to_bf16_kernel<<<2048, 256, 0, stream>>>(decW, wdb, NDEC);

    embed_kernel<<<BT, 256, 0, stream>>>(ids, emb, pos, x, xb);

    const long TT  = (long)Tc * Tc;
    const long TCs = (long)Tc * Cc;

    const dim3 gFull(Cc / 64, BT / 128, 1);
    const dim3 gAtt (Tc / 64, Tc / 128, Bc * Hc);
    const dim3 gSmV (Dc / 64, Tc / 128, Bc * Hc);
    const dim3 gFF1 (FC / 64, BT / 128, 1);
    const dim3 gDec (Vc / 64, BT / 128, 1);

    for (int l = 0; l < Lc; ++l) {
        const __bf16* wq_l = wqb + (long)l * Hc * Cc * Dc;
        const __bf16* wk_l = wkb + (long)l * Hc * Cc * Dc;
        const __bf16* wv_l = wvb + (long)l * Hc * Cc * Dc;
        const __bf16* wo_l = wob + (long)l * Cc * Cc;
        const __bf16* w1_l = w1b + (long)l * Cc * FC;
        const __bf16* w2_l = w2b + (long)l * FC * Cc;
        const float* bo_l = bo + (long)l * Cc;
        const float* b1_l = b1 + (long)l * FC;
        const float* b2_l = b2 + (long)l * Cc;

        // Q/K/V -> [b,t,h,d] bf16
        gemm_wmma_bf16<<<gFull, 128, 0, stream>>>(xb, Cc, wq_l, 64, 2, nullptr,
            1.0f, 0, nullptr, Qb, Cc, Cc, 1, 0, 0, 0, 0, 0, 0);
        gemm_wmma_bf16<<<gFull, 128, 0, stream>>>(xb, Cc, wk_l, 64, 2, nullptr,
            1.0f, 0, nullptr, Kb, Cc, Cc, 1, 0, 0, 0, 0, 0, 0);
        gemm_wmma_bf16<<<gFull, 128, 0, stream>>>(xb, Cc, wv_l, 64, 2, nullptr,
            1.0f, 0, nullptr, Vb, Cc, Cc, 1, 0, 0, 0, 0, 0, 0);

        // att[b,h,q,k] = (Q.K)/8, batched over 96 (b,h), B transposed
        gemm_wmma_bf16<<<gAtt, 128, 0, stream>>>(Qb, Cc, Kb, Cc, 1, nullptr,
            0.125f, 0, attf, nullptr, Tc, Dc, Hc,
            TCs, Dc, TCs, Dc, (long)Hc * TT, TT);

        // softmax over query axis -> bf16
        softmax_q_axis<<<dim3(Tc / 256, Bc * Hc), 256, 0, stream>>>(attf, attb);

        // O[b,q,h,d] = sm @ V (batched, N=64) -> bf16
        gemm_wmma_bf16<<<gSmV, 128, 0, stream>>>(attb, Tc, Vb, Cc, 0, nullptr,
            1.0f, 0, nullptr, Ob, Cc, Tc, Hc,
            (long)Hc * TT, TT, TCs, Dc, TCs, Dc);

        // Wo projection + bias -> t1 (f32)
        gemm_wmma_bf16<<<gFull, 128, 0, stream>>>(Ob, Cc, wo_l, Cc, 0, bo_l,
            1.0f, 0, t1, nullptr, Cc, Cc, 1, 0, 0, 0, 0, 0, 0);

        // x = LN(x + attn_out)
        add_layernorm<<<BT, 256, 0, stream>>>(x, t1, ln1g + (long)l * Cc,
                                              ln1b + (long)l * Cc, x, xb);

        // FFN
        gemm_wmma_bf16<<<gFF1, 128, 0, stream>>>(xb, Cc, w1_l, FC, 0, b1_l,
            1.0f, 1, nullptr, h1b, FC, Cc, 1, 0, 0, 0, 0, 0, 0);
        gemm_wmma_bf16<<<gFull, 128, 0, stream>>>(h1b, FC, w2_l, Cc, 0, b2_l,
            1.0f, 0, t1, nullptr, Cc, FC, 1, 0, 0, 0, 0, 0, 0);

        // x = LN(x + ffn_out)
        add_layernorm<<<BT, 256, 0, stream>>>(x, t1, ln2g + (long)l * Cc,
                                              ln2b + (long)l * Cc, x, xb);
    }

    // Decoder: out[BT,V] = x @ dec_W + dec_b  (~201 GFLOP)
    gemm_wmma_bf16<<<gDec, 128, 0, stream>>>(xb, Cc, wdb, Vc, 0, decb,
        1.0f, 0, out, nullptr, Vc, Cc, 1, 0, 0, 0, 0, 0, 0);
}
